// HierarchicalMOE_57621281243507
// MI455X (gfx1250) — compile-verified, matrix operand order checked by reading, not run on hardware
//
#include <hip/hip_runtime.h>
#include <hip/hip_bf16.h>
#include <math.h>

typedef __bf16 bf16;
typedef __attribute__((ext_vector_type(16))) __bf16 v16bf;
typedef __attribute__((ext_vector_type(8)))  float  v8f;

#define NTOK 2048
#define DIM  768
#define HID  3072
#define NGRP 4
#define NEXP 4
#define NE   16
#define LN_EPS 1e-5f

// ---------------------------------------------------------------- helpers
static __device__ __forceinline__ float gelu_exact(float v) {
    return 0.5f * v * (1.0f + erff(v * 0.70710678118654752f));
}

// CDNA5 async global->LDS copy: 16 bytes per lane, tracked on ASYNCcnt.
static __device__ __forceinline__ void async_copy16(void* lds_dst, const void* gsrc) {
    unsigned loff = (unsigned)(size_t)lds_dst;             // addr[31:0] = LDS offset
    unsigned long long ga = (unsigned long long)(size_t)gsrc;
    asm volatile("global_load_async_to_lds_b128 %0, %1, off"
                 :: "v"(loff), "v"(ga) : "memory");
}
// Wait until at most N async ops remain in flight (in-order completion =>
// all older tile loads have landed).
template <int N>
static __device__ __forceinline__ void wait_async_n() {
    asm volatile("s_wait_asynccnt %0" :: "n"(N) : "memory");
}

// Build a 16-row x 32-K bf16 WMMA fragment from an LDS row chunk.
// ISA 16-bit layout: lane half h: elems 0..7 -> K=h*8+j; elems 8..15 -> K=16+h*8+(j-8).
static __device__ __forceinline__ v16bf load_frag(const bf16* rowp, int half) {
    v16bf f;
    const bf16* p0 = rowp + half * 8;
    const bf16* p1 = rowp + 16 + half * 8;
#pragma unroll
    for (int j = 0; j < 8; ++j) { f[j] = p0[j]; f[8 + j] = p1[j]; }
    return f;
}

// ---------------------------------------------------------------- fp32 -> bf16 transpose
// src: [Ksz][Nsz] f32 (row-major), dst: [Nsz][Ksz] bf16. blockIdx.z = batch.
__global__ __launch_bounds__(256) void transpose_cvt_kernel(const float* __restrict__ src,
                                                            bf16* __restrict__ dst,
                                                            int Ksz, int Nsz) {
    src += (size_t)blockIdx.z * Ksz * Nsz;
    dst += (size_t)blockIdx.z * Ksz * Nsz;
    int k0 = blockIdx.y * 32, n0 = blockIdx.x * 32;
    int t = threadIdx.x;
    __shared__ float tile[32][36];

    int r = t >> 3, c4 = (t & 7) * 4;
    float4 v = *reinterpret_cast<const float4*>(&src[(size_t)(k0 + r) * Nsz + n0 + c4]);
    tile[r][c4 + 0] = v.x; tile[r][c4 + 1] = v.y; tile[r][c4 + 2] = v.z; tile[r][c4 + 3] = v.w;
    __syncthreads();

    union { bf16 h[4]; uint2 u; } pk;
#pragma unroll
    for (int j = 0; j < 4; ++j) pk.h[j] = (bf16)tile[c4 + j][r];
    *reinterpret_cast<uint2*>(&dst[(size_t)(n0 + r) * Ksz + k0 + c4]) = pk.u;
}

// ---------------------------------------------------------------- routing
__global__ __launch_bounds__(128) void route_kernel(const float* __restrict__ x,
                                                    const float* __restrict__ Wg,
                                                    const float* __restrict__ bg,
                                                    const float* __restrict__ We,
                                                    const float* __restrict__ be,
                                                    bf16* __restrict__ xbf,
                                                    int* __restrict__ sel_ge,
                                                    float* __restrict__ sel_w) {
    int wave = threadIdx.x >> 5, lane = threadIdx.x & 31;
    int t = blockIdx.x * 4 + wave;
    if (t >= NTOK) return;

    float xr[24];
#pragma unroll
    for (int i = 0; i < 24; ++i) {
        int d = lane + 32 * i;
        xr[i] = x[(size_t)t * DIM + d];
        xbf[(size_t)t * DIM + d] = (bf16)xr[i];
    }
    float acc[20];
#pragma unroll
    for (int j = 0; j < 20; ++j) acc[j] = 0.f;
    for (int i = 0; i < 24; ++i) {
        int d = lane + 32 * i;
        float xv = xr[i];
#pragma unroll
        for (int g = 0; g < NGRP; ++g) acc[g] += xv * Wg[d * NGRP + g];
#pragma unroll
        for (int g = 0; g < NGRP; ++g)
#pragma unroll
            for (int e = 0; e < NEXP; ++e)
                acc[4 + g * 4 + e] += xv * We[((size_t)g * DIM + d) * NEXP + e];
    }
#pragma unroll
    for (int j = 0; j < 20; ++j)
#pragma unroll
        for (int off = 16; off > 0; off >>= 1) acc[j] += __shfl_xor(acc[j], off, 32);

    if (lane == 0) {
        float lg[4], pg[4];
#pragma unroll
        for (int g = 0; g < 4; ++g) lg[g] = acc[g] + bg[g];
        float mx = lg[0];
#pragma unroll
        for (int g = 1; g < 4; ++g) mx = fmaxf(mx, lg[g]);
        float sum = 0.f;
#pragma unroll
        for (int g = 0; g < 4; ++g) { pg[g] = __expf(lg[g] - mx); sum += pg[g]; }
#pragma unroll
        for (int g = 0; g < 4; ++g) pg[g] /= sum;
        int i0 = 0;
#pragma unroll
        for (int g = 1; g < 4; ++g) if (pg[g] > pg[i0]) i0 = g;
        int i1 = (i0 == 0) ? 1 : 0;
#pragma unroll
        for (int g = 0; g < 4; ++g) if (g != i0 && pg[g] > pg[i1]) i1 = g;

#pragma unroll
        for (int k = 0; k < 2; ++k) {
            int g = (k == 0) ? i0 : i1;
            float le[4], pe[4];
#pragma unroll
            for (int e = 0; e < 4; ++e) le[e] = acc[4 + g * 4 + e] + be[g * 4 + e];
            float m2 = le[0];
#pragma unroll
            for (int e = 1; e < 4; ++e) m2 = fmaxf(m2, le[e]);
            float s2 = 0.f;
#pragma unroll
            for (int e = 0; e < 4; ++e) { pe[e] = __expf(le[e] - m2); s2 += pe[e]; }
            int eb = 0;
#pragma unroll
            for (int e = 1; e < 4; ++e) if (pe[e] > pe[eb]) eb = e;
            sel_ge[t * 2 + k] = g * 4 + eb;
            sel_w[t * 2 + k] = pg[g] * (pe[eb] / s2);
        }
    }
}

// ---------------------------------------------------------------- token lists (deterministic)
__global__ void build_lists_kernel(const int* __restrict__ sel_ge,
                                   const float* __restrict__ sel_w,
                                   int* __restrict__ counts,
                                   int* __restrict__ ltok,
                                   int* __restrict__ lslot,
                                   float* __restrict__ lw) {
    int e = threadIdx.x;
    if (e >= NE) return;
    int c = 0;
    for (int t = 0; t < NTOK; ++t)
        for (int k = 0; k < 2; ++k)
            if (sel_ge[t * 2 + k] == e) {
                ltok[e * NTOK + c] = t;
                lslot[e * NTOK + c] = k;
                lw[e * NTOK + c] = sel_w[t * 2 + k];
                ++c;
            }
    counts[e] = c;
}

// ---------------------------------------------------------------- GEMM1: h = gelu(Xe @ W1 + b1)
// W1T: [e][h][d] bf16. Tile M=32 x N=128, K-step 64, double-buffered B.
__global__ __launch_bounds__(256) void ffn1_kernel(const bf16* __restrict__ xbf,
                                                   const bf16* __restrict__ W1T,
                                                   const float* __restrict__ b1,
                                                   const int* __restrict__ counts,
                                                   const int* __restrict__ ltok,
                                                   bf16* __restrict__ hbf) {
    int e = blockIdx.z;
    int cnt = counts[e];
    int row0 = blockIdx.y * 32;
    if (row0 >= cnt) return;
    int nc0 = blockIdx.x * 128;
    int t = threadIdx.x;

    __shared__ __align__(16) bf16 Ash[32][DIM + 8];     // full-K A tile (gathered once)
    __shared__ __align__(16) bf16 Bsh[2][128][72];      // double-buffered B tiles

    {   // gather A tile (vectorized 16B copies)
        int r = t >> 3, base = (t & 7) * 96;
        int ridx = row0 + r; if (ridx >= cnt) ridx = cnt - 1;
        int tok = ltok[e * NTOK + ridx];
        const bf16* src = xbf + (size_t)tok * DIM + base;
        bf16* dst = &Ash[r][base];
#pragma unroll
        for (int i = 0; i < 12; ++i)
            *reinterpret_cast<uint4*>(dst + i * 8) =
                *reinterpret_cast<const uint4*>(src + i * 8);
    }

    v8f acc0 = {}, acc1 = {};
    int lane = t & 31, w = t >> 5, half = lane >> 4, nl = lane & 15;
    int bn = t >> 1, bq = (t & 1) * 32;
    const bf16* brow = W1T + ((size_t)e * HID + nc0 + bn) * DIM + bq;

    const int KT = DIM / 64;
    // prologue: tile 0 -> buf 0 (4 async ops per wave)
#pragma unroll
    for (int i = 0; i < 4; ++i)
        async_copy16(&Bsh[0][bn][bq + i * 8], brow + i * 8);

    for (int kk = 0; kk < KT; ++kk) {
        if (kk + 1 < KT) {          // issue next tile into alternate buffer
#pragma unroll
            for (int i = 0; i < 4; ++i)
                async_copy16(&Bsh[(kk + 1) & 1][bn][bq + i * 8],
                             brow + (kk + 1) * 64 + i * 8);
            wait_async_n<4>();      // tile kk landed; kk+1 still in flight
        } else {
            wait_async_n<0>();
        }
        __syncthreads();
        const bf16(*Bc)[72] = Bsh[kk & 1];
#pragma unroll
        for (int s = 0; s < 2; ++s) {
            v16bf bF = load_frag(&Bc[w * 16 + nl][s * 32], half);
            v16bf a0 = load_frag(&Ash[nl][kk * 64 + s * 32], half);
            v16bf a1 = load_frag(&Ash[16 + nl][kk * 64 + s * 32], half);
            acc0 = __builtin_amdgcn_wmma_f32_16x16x32_bf16(false, a0, false, bF, (short)0, acc0, false, false);
            acc1 = __builtin_amdgcn_wmma_f32_16x16x32_bf16(false, a1, false, bF, (short)0, acc1, false, false);
        }
        __syncthreads();            // all waves done reading buf kk&1
    }

    int ncol = nc0 + w * 16 + nl;
    float bias = b1[e * HID + ncol];
#pragma unroll
    for (int r = 0; r < 8; ++r) {
        int m = half * 8 + r;
        int row = row0 + m;
        if (row < cnt)
            hbf[((size_t)e * NTOK + row) * HID + ncol] = (bf16)gelu_exact(acc0[r] + bias);
        row = row0 + 16 + m;
        if (row < cnt)
            hbf[((size_t)e * NTOK + row) * HID + ncol] = (bf16)gelu_exact(acc1[r] + bias);
    }
}

// ---------------------------------------------------------------- GEMM2: slots = (h @ W2 + b2) * w
// W2T: [e][f][k] bf16 (k over HID). Double-buffered A+B.
__global__ __launch_bounds__(256) void ffn2_kernel(const bf16* __restrict__ hbf,
                                                   const bf16* __restrict__ W2T,
                                                   const float* __restrict__ b2,
                                                   const int* __restrict__ counts,
                                                   const int* __restrict__ ltok,
                                                   const int* __restrict__ lslot,
                                                   const float* __restrict__ lw,
                                                   float* __restrict__ slots) {
    int e = blockIdx.z;
    int cnt = counts[e];
    int row0 = blockIdx.y * 32;
    if (row0 >= cnt) return;
    int nc0 = blockIdx.x * 128;
    int t = threadIdx.x;

    __shared__ __align__(16) bf16 Ash[2][32][72];
    __shared__ __align__(16) bf16 Bsh[2][128][72];

    v8f acc0 = {}, acc1 = {};
    int lane = t & 31, w = t >> 5, half = lane >> 4, nl = lane & 15;
    int ar = t >> 3, ac = (t & 7) * 8;
    int bn = t >> 1, bq = (t & 1) * 32;
    const bf16* arow = hbf + ((size_t)e * NTOK + row0 + ar) * HID + ac;
    const bf16* brow = W2T + ((size_t)e * DIM + nc0 + bn) * HID + bq;

    const int KT = HID / 64;
    async_copy16(&Ash[0][ar][ac], arow);
#pragma unroll
    for (int i = 0; i < 4; ++i)
        async_copy16(&Bsh[0][bn][bq + i * 8], brow + i * 8);

    for (int kk = 0; kk < KT; ++kk) {
        if (kk + 1 < KT) {
            int nb = (kk + 1) & 1;
            async_copy16(&Ash[nb][ar][ac], arow + (kk + 1) * 64);
#pragma unroll
            for (int i = 0; i < 4; ++i)
                async_copy16(&Bsh[nb][bn][bq + i * 8], brow + (kk + 1) * 64 + i * 8);
            wait_async_n<5>();
        } else {
            wait_async_n<0>();
        }
        __syncthreads();
        const bf16(*Ac)[72] = Ash[kk & 1];
        const bf16(*Bc)[72] = Bsh[kk & 1];
#pragma unroll
        for (int s = 0; s < 2; ++s) {
            v16bf bF = load_frag(&Bc[w * 16 + nl][s * 32], half);
            v16bf a0 = load_frag(&Ac[nl][s * 32], half);
            v16bf a1 = load_frag(&Ac[16 + nl][s * 32], half);
            acc0 = __builtin_amdgcn_wmma_f32_16x16x32_bf16(false, a0, false, bF, (short)0, acc0, false, false);
            acc1 = __builtin_amdgcn_wmma_f32_16x16x32_bf16(false, a1, false, bF, (short)0, acc1, false, false);
        }
        __syncthreads();
    }

    int ncol = nc0 + w * 16 + nl;
    float bias = b2[e * DIM + ncol];
#pragma unroll
    for (int r = 0; r < 8; ++r) {
        int m = half * 8 + r;
#pragma unroll
        for (int p = 0; p < 2; ++p) {
            int row = row0 + p * 16 + m;
            if (row < cnt) {
                int li = e * NTOK + row;
                float v = ((p ? acc1[r] : acc0[r]) + bias) * lw[li];
                slots[((size_t)ltok[li] * 2 + lslot[li]) * DIM + ncol] = v;
            }
        }
    }
}

// ---------------------------------------------------------------- combine slots -> bf16
__global__ __launch_bounds__(256) void combine_kernel(const float* __restrict__ slots,
                                                      bf16* __restrict__ ybf) {
    int i = blockIdx.x * 256 + threadIdx.x;
    int tk = i / DIM, d = i - tk * DIM;
    float v = slots[(size_t)tk * (2 * DIM) + d] + slots[(size_t)tk * (2 * DIM) + DIM + d];
    ybf[i] = (bf16)v;
}

// ---------------------------------------------------------------- GEMM3: yp = y @ Wp + bp
// WpT: [f][d] bf16. Double-buffered A+B.
__global__ __launch_bounds__(256) void proj_kernel(const bf16* __restrict__ ybf,
                                                   const bf16* __restrict__ WpT,
                                                   const float* __restrict__ bp,
                                                   float* __restrict__ yp) {
    int row0 = blockIdx.y * 32;
    int nc0 = blockIdx.x * 128;
    int t = threadIdx.x;

    __shared__ __align__(16) bf16 Ash[2][32][72];
    __shared__ __align__(16) bf16 Bsh[2][128][72];

    v8f acc0 = {}, acc1 = {};
    int lane = t & 31, w = t >> 5, half = lane >> 4, nl = lane & 15;
    int ar = t >> 3, ac = (t & 7) * 8;
    int bn = t >> 1, bq = (t & 1) * 32;
    const bf16* arow = ybf + (size_t)(row0 + ar) * DIM + ac;
    const bf16* brow = WpT + (size_t)(nc0 + bn) * DIM + bq;

    const int KT = DIM / 64;
    async_copy16(&Ash[0][ar][ac], arow);
#pragma unroll
    for (int i = 0; i < 4; ++i)
        async_copy16(&Bsh[0][bn][bq + i * 8], brow + i * 8);

    for (int kk = 0; kk < KT; ++kk) {
        if (kk + 1 < KT) {
            int nb = (kk + 1) & 1;
            async_copy16(&Ash[nb][ar][ac], arow + (kk + 1) * 64);
#pragma unroll
            for (int i = 0; i < 4; ++i)
                async_copy16(&Bsh[nb][bn][bq + i * 8], brow + (kk + 1) * 64 + i * 8);
            wait_async_n<5>();
        } else {
            wait_async_n<0>();
        }
        __syncthreads();
        const bf16(*Ac)[72] = Ash[kk & 1];
        const bf16(*Bc)[72] = Bsh[kk & 1];
#pragma unroll
        for (int s = 0; s < 2; ++s) {
            v16bf bF = load_frag(&Bc[w * 16 + nl][s * 32], half);
            v16bf a0 = load_frag(&Ac[nl][s * 32], half);
            v16bf a1 = load_frag(&Ac[16 + nl][s * 32], half);
            acc0 = __builtin_amdgcn_wmma_f32_16x16x32_bf16(false, a0, false, bF, (short)0, acc0, false, false);
            acc1 = __builtin_amdgcn_wmma_f32_16x16x32_bf16(false, a1, false, bF, (short)0, acc1, false, false);
        }
        __syncthreads();
    }

    int ncol = nc0 + w * 16 + nl;
    float bias = bp[ncol];
#pragma unroll
    for (int r = 0; r < 8; ++r) {
        int m = half * 8 + r;
        yp[(size_t)(row0 + m) * DIM + ncol] = acc0[r] + bias;
        yp[(size_t)(row0 + 16 + m) * DIM + ncol] = acc1[r] + bias;
    }
}

// ---------------------------------------------------------------- LayerNorm (two-pass)
__global__ __launch_bounds__(256) void ln_kernel(const float* __restrict__ yp,
                                                 const float* __restrict__ gamma,
                                                 const float* __restrict__ beta,
                                                 float* __restrict__ out) {
    int tk = blockIdx.x;
    const float* row = yp + (size_t)tk * DIM;
    int t = threadIdx.x;
    __shared__ float red[256];

    float s = 0.f;
#pragma unroll
    for (int i = 0; i < 3; ++i) s += row[t + 256 * i];
    red[t] = s; __syncthreads();
    for (int o = 128; o > 0; o >>= 1) { if (t < o) red[t] += red[t + o]; __syncthreads(); }
    float mu = red[0] / (float)DIM;
    __syncthreads();

    float s2 = 0.f;
#pragma unroll
    for (int i = 0; i < 3; ++i) { float d = row[t + 256 * i] - mu; s2 += d * d; }
    red[t] = s2; __syncthreads();
    for (int o = 128; o > 0; o >>= 1) { if (t < o) red[t] += red[t + o]; __syncthreads(); }
    float inv = rsqrtf(red[0] / (float)DIM + LN_EPS);

#pragma unroll
    for (int i = 0; i < 3; ++i) {
        int d = t + 256 * i;
        out[(size_t)tk * DIM + d] = (row[d] - mu) * inv * gamma[d] + beta[d];
    }
}

// ---------------------------------------------------------------- launch
extern "C" void kernel_launch(void* const* d_in, const int* in_sizes, int n_in,
                              void* d_out, int out_size, void* d_ws, size_t ws_size,
                              hipStream_t stream) {
    (void)in_sizes; (void)n_in; (void)out_size; (void)ws_size;
    const float* x     = (const float*)d_in[0];
    const float* Wg    = (const float*)d_in[1];
    const float* bg    = (const float*)d_in[2];
    const float* We    = (const float*)d_in[3];
    const float* be    = (const float*)d_in[4];
    const float* W1    = (const float*)d_in[5];
    const float* b1    = (const float*)d_in[6];
    const float* W2    = (const float*)d_in[7];
    const float* b2    = (const float*)d_in[8];
    const float* Wp    = (const float*)d_in[9];
    const float* bp    = (const float*)d_in[10];
    const float* gamma = (const float*)d_in[11];
    const float* beta  = (const float*)d_in[12];
    float* out = (float*)d_out;

    char* w = (char*)d_ws;
    size_t off = 0;
    auto take = [&](size_t bytes) -> void* {
        void* p = w + off;
        off += (bytes + 255) & ~(size_t)255;
        return p;
    };
    bf16*  W1T    = (bf16*)take((size_t)NE * DIM * HID * sizeof(bf16));
    bf16*  W2T    = (bf16*)take((size_t)NE * HID * DIM * sizeof(bf16));
    bf16*  WpT    = (bf16*)take((size_t)DIM * DIM * sizeof(bf16));
    bf16*  xbf    = (bf16*)take((size_t)NTOK * DIM * sizeof(bf16));
    bf16*  hbf    = (bf16*)take((size_t)NE * NTOK * HID * sizeof(bf16));
    float* slots  = (float*)take((size_t)NTOK * 2 * DIM * sizeof(float));
    bf16*  ybf    = (bf16*)take((size_t)NTOK * DIM * sizeof(bf16));
    float* yp     = (float*)take((size_t)NTOK * DIM * sizeof(float));
    int*   sel_ge = (int*)take((size_t)NTOK * 2 * sizeof(int));
    float* sel_w  = (float*)take((size_t)NTOK * 2 * sizeof(float));
    int*   counts = (int*)take(NE * sizeof(int));
    int*   ltok   = (int*)take((size_t)NE * NTOK * sizeof(int));
    int*   lslot  = (int*)take((size_t)NE * NTOK * sizeof(int));
    float* lw     = (float*)take((size_t)NE * NTOK * sizeof(float));

    // weight convert + transpose to [N][K] bf16
    transpose_cvt_kernel<<<dim3(HID / 32, DIM / 32, NE), 256, 0, stream>>>(W1, W1T, DIM, HID);
    transpose_cvt_kernel<<<dim3(DIM / 32, HID / 32, NE), 256, 0, stream>>>(W2, W2T, HID, DIM);
    transpose_cvt_kernel<<<dim3(DIM / 32, DIM / 32, 1), 256, 0, stream>>>(Wp, WpT, DIM, DIM);

    // routing + x bf16
    route_kernel<<<NTOK / 4, 128, 0, stream>>>(x, Wg, bg, We, be, xbf, sel_ge, sel_w);

    // deterministic per-expert token lists
    build_lists_kernel<<<1, NE, 0, stream>>>(sel_ge, sel_w, counts, ltok, lslot, lw);

    // sparse expert FFN
    ffn1_kernel<<<dim3(HID / 128, NTOK / 32, NE), 256, 0, stream>>>(xbf, W1T, b1, counts, ltok, hbf);
    ffn2_kernel<<<dim3(DIM / 128, NTOK / 32, NE), 256, 0, stream>>>(hbf, W2T, b2, counts, ltok, lslot, lw, slots);

    // combine + projection + layernorm
    combine_kernel<<<NTOK * DIM / 256, 256, 0, stream>>>(slots, ybf);
    proj_kernel<<<dim3(DIM / 128, NTOK / 32), 256, 0, stream>>>(ybf, WpT, bp, yp);
    ln_kernel<<<NTOK, 256, 0, stream>>>(yp, gamma, beta, out);
}
